// TransformerMAX_7756710936727
// MI455X (gfx1250) — compile-verified, matrix operand order checked by reading, not run on hardware
//
#include <hip/hip_runtime.h>

// ---------------- model dimensions ----------------
#define LNUM    2
#define DMODEL  512
#define NHEAD   8
#define DHEAD   64
#define FFDIM   2048
#define VOCAB   32000
#define SEQ     256
#define BATCH   2
#define MEMSL   128
#define MAXLEN  256
#define RELPOS  (2 * MAXLEN - 1)

typedef __attribute__((ext_vector_type(8)))  float  v8f;
typedef __attribute__((ext_vector_type(16))) __bf16 v16bf;
typedef __attribute__((ext_vector_type(8)))  __bf16 v8bf;
typedef int v4i __attribute__((vector_size(16)));

union bfrag { v16bf v; v8bf h[2]; };

// gfx1250 has native bf16 converts -> let clang pick v_cvt (packed where possible)
__device__ __forceinline__ __bf16 f2bf(float f) { return (__bf16)f; }

#if defined(__AMDGCN__) && __has_builtin(__builtin_amdgcn_global_load_async_to_lds_b128)
#define HAS_ASYNC_LDS 1
#else
#define HAS_ASYNC_LDS 0
#endif

#define AS1G __attribute__((address_space(1)))
#define AS3L __attribute__((address_space(3)))

__device__ __forceinline__ void wait_async0() {
#if defined(__AMDGCN__)
#if __has_builtin(__builtin_amdgcn_s_wait_asynccnt)
  __builtin_amdgcn_s_wait_asynccnt(0);
#else
  asm volatile("s_wait_asynccnt 0" ::: "memory");
#endif
#endif
}

// ---------------- WMMA GEMM ----------------
// C = outScale * ( R + relu( alpha*(A @ opB) + bias ) )
//   A: MxK f32 row-major.
//   B bBf16=1: NxK row-major bf16 (pre-converted weights; async global->LDS)
//   B bBf16=0, bT=1: NxK row-major f32 (activation K^T)
//   B bBf16=0, bT=0: KxN row-major f32 (plain matmul; transpose while staging)
#define BM   128
#define BN   64
#define BK   32
#define LDAP 40   // padded LDS row stride (bf16) -> 16B aligned b128 frags
#define LDBP 40

struct GemmP {
  const float* A; long lda, sA1, sA2;
  const void*  B; long ldb, sB1, sB2;
  const float* bias;
  const float* R; long ldr, sR1, sR2;
  float* C;       long ldc, sC1, sC2;
  int nb1;
  int K;
  float alpha, outScale;
  int bT, relu, bBf16;
};

__global__ __launch_bounds__(256) void gemm_wmma_k(GemmP p) {
  __shared__ __bf16 sA[2][BM * LDAP];
  __shared__ __bf16 sB[2][BN * LDBP];

  const int z  = blockIdx.z;
  const int b1 = z % p.nb1, b2 = z / p.nb1;
  const long bOff = (long)b1 * p.sB1 + (long)b2 * p.sB2;
  const float* Ag = p.A + (long)b1 * p.sA1 + (long)b2 * p.sA2 +
                    (long)blockIdx.x * BM * p.lda;
  const int tileN = blockIdx.y * BN;
  const int tid   = threadIdx.x;

  auto stage = [&](int buf, int k0) {
    // A tile: 128 x 32 f32 -> bf16 (1024 float4 slots)
#pragma unroll
    for (int it = 0; it < 4; ++it) {
      int slot = tid + it * 256;
      int row = slot >> 3, c4 = (slot & 7) * 4;
      float4 f = *(const float4*)(Ag + (long)row * p.lda + k0 + c4);
      __bf16* d = &sA[buf][row * LDAP + c4];
      d[0] = f2bf(f.x); d[1] = f2bf(f.y); d[2] = f2bf(f.z); d[3] = f2bf(f.w);
    }
    if (p.bBf16) {
      // B tile: 64 x 32 bf16, NT. 256 chunks of 8 bf16 (16B) -> 1 per thread.
      int row = tid >> 2, c8 = (tid & 3) * 8;
      const __bf16* gp = (const __bf16*)p.B + bOff +
                         (long)(tileN + row) * p.ldb + k0 + c8;
      __bf16* lp = &sB[buf][row * LDBP + c8];
#if HAS_ASYNC_LDS
      __builtin_amdgcn_global_load_async_to_lds_b128(
          (AS1G v4i*)(v4i*)gp, (AS3L v4i*)(v4i*)lp, 0, 0);
#else
      *(v8bf*)lp = *(const v8bf*)gp;
#endif
    } else if (p.bT) {            // B NxK f32 row-major
      const float* Bg = (const float*)p.B + bOff;
#pragma unroll
      for (int it = 0; it < 2; ++it) {
        int slot = tid + it * 256;                    // 512 slots
        int row = slot >> 3, c4 = (slot & 7) * 4;
        float4 f = *(const float4*)(Bg + (long)(tileN + row) * p.ldb + k0 + c4);
        __bf16* d = &sB[buf][row * LDBP + c4];
        d[0] = f2bf(f.x); d[1] = f2bf(f.y); d[2] = f2bf(f.z); d[3] = f2bf(f.w);
      }
    } else {                      // B KxN f32 row-major: transpose while staging
      const float* Bg = (const float*)p.B + bOff;
#pragma unroll
      for (int it = 0; it < 2; ++it) {
        int slot = tid + it * 256;                    // 32 k-rows x 16 n4
        int k = slot >> 4, n4 = (slot & 15) * 4;
        float4 f = *(const float4*)(Bg + (long)(k0 + k) * p.ldb + tileN + n4);
        sB[buf][(n4 + 0) * LDBP + k] = f2bf(f.x);
        sB[buf][(n4 + 1) * LDBP + k] = f2bf(f.y);
        sB[buf][(n4 + 2) * LDBP + k] = f2bf(f.z);
        sB[buf][(n4 + 3) * LDBP + k] = f2bf(f.w);
      }
    }
  };

  const int wave = tid >> 5, lane = tid & 31;
  const int wm = wave & 3, wn = wave >> 2;    // 4 x 2 waves -> 32x32 per wave
  const int hl = lane >> 4, l16 = lane & 15;

  v8f acc[2][2] = {};

  stage(0, 0);
#if HAS_ASYNC_LDS
  if (p.bBf16) wait_async0();
#endif
  __syncthreads();
  const int nk = p.K / BK;
  for (int ks = 0; ks < nk; ++ks) {
    const int buf = ks & 1;
    if (ks + 1 < nk) stage(buf ^ 1, (ks + 1) * BK);
    if (ks + 2 < nk)   // pull K+2 A tile toward L2/L0 (global_prefetch_b8)
      __builtin_prefetch(Ag + (long)(tid >> 1) * p.lda + (ks + 2) * BK, 0, 1);

    bfrag a[2], b[2];
    // A 16x32 bf16 lane layout: lanes 0-15 -> K 0-7 & 16-23, lanes 16-31 -> K 8-15 & 24-31
#pragma unroll
    for (int mi = 0; mi < 2; ++mi) {
      const __bf16* ptr = &sA[buf][(wm * 32 + mi * 16 + l16) * LDAP + hl * 8];
      a[mi].h[0] = *(const v8bf*)ptr;
      a[mi].h[1] = *(const v8bf*)(ptr + 16);
    }
#pragma unroll
    for (int ni = 0; ni < 2; ++ni) {
      const __bf16* ptr = &sB[buf][(wn * 32 + ni * 16 + l16) * LDBP + hl * 8];
      b[ni].h[0] = *(const v8bf*)ptr;
      b[ni].h[1] = *(const v8bf*)(ptr + 16);
    }
#pragma unroll
    for (int mi = 0; mi < 2; ++mi)
#pragma unroll
      for (int ni = 0; ni < 2; ++ni)
        acc[mi][ni] = __builtin_amdgcn_wmma_f32_16x16x32_bf16(
            false, a[mi].v, false, b[ni].v, (short)0, acc[mi][ni], false, false);
#if HAS_ASYNC_LDS
    if (p.bBf16) wait_async0();
#endif
    __syncthreads();
  }

  // epilogue: C/D layout -> VGPR r holds row (hl*8 + r), col = l16
  float* Cg = p.C + (long)b1 * p.sC1 + (long)b2 * p.sC2 +
              (long)blockIdx.x * BM * p.ldc;
  const float* Rg = p.R ? p.R + (long)b1 * p.sR1 + (long)b2 * p.sR2 +
                          (long)blockIdx.x * BM * p.ldr
                        : nullptr;
#pragma unroll
  for (int mi = 0; mi < 2; ++mi) {
#pragma unroll
    for (int ni = 0; ni < 2; ++ni) {
      const int col = tileN + wn * 32 + ni * 16 + l16;
      const float bb = p.bias ? p.bias[col] : 0.f;
#pragma unroll
      for (int r = 0; r < 8; ++r) {
        const int row = wm * 32 + mi * 16 + hl * 8 + r;
        float v = acc[mi][ni][r] * p.alpha + bb;
        if (p.relu) v = fmaxf(v, 0.f);
        if (Rg) v += Rg[(long)row * p.ldr + col];
        v *= p.outScale;
        Cg[(long)row * p.ldc + col] = v;
      }
    }
  }
}

// ---------------- auxiliary kernels ----------------
__global__ void cvt_bf16_k(const float* in, __bf16* out, long n) {
  const long i = ((long)blockIdx.x * blockDim.x + threadIdx.x) * 4;
  if (i + 3 < n) {
    float4 f = *(const float4*)(in + i);
    out[i + 0] = f2bf(f.x); out[i + 1] = f2bf(f.y);
    out[i + 2] = f2bf(f.z); out[i + 3] = f2bf(f.w);
  } else {
    for (long j = i; j < n; ++j) out[j] = f2bf(in[j]);
  }
}

__global__ void embed_k(const int* ids, const float* tok, const float* pos,
                        float* out) {
  const int row = blockIdx.x;              // over B*S
  const int s = row % SEQ;
  const int id = ids[row];
  const long ob = (long)row * DMODEL;
  for (int j = threadIdx.x; j < DMODEL; j += blockDim.x)
    out[ob + j] = tok[(long)id * DMODEL + j] + pos[(long)s * DMODEL + j];
}

__global__ __launch_bounds__(256) void ln_k(const float* x, const float* res,
                                            float* y) {   // row length == 512
  __shared__ float s1[256], s2[256];
  const long base = (long)blockIdx.x * DMODEL;
  const int t = threadIdx.x;
  float a = x[base + t]       + (res ? res[base + t]       : 0.f);
  float b = x[base + t + 256] + (res ? res[base + t + 256] : 0.f);
  s1[t] = a + b; s2[t] = a * a + b * b;
  __syncthreads();
  for (int o = 128; o > 0; o >>= 1) {
    if (t < o) { s1[t] += s1[t + o]; s2[t] += s2[t + o]; }
    __syncthreads();
  }
  const float mean = s1[0] * (1.f / DMODEL);
  const float var  = s2[0] * (1.f / DMODEL) - mean * mean;
  const float rs = rsqrtf(var + 1e-5f);
  y[base + t]       = (a - mean) * rs;
  y[base + t + 256] = (b - mean) * rs;
}

__global__ __launch_bounds__(256) void softmax_k(float* x, int cols, int sq,
                                                 int causal) {
  __shared__ float sh[256];
  const long row = blockIdx.x;
  const long base = row * cols;
  const int t = threadIdx.x;
  const int q = (int)(row % sq);
  const bool act = t < cols;
  float val = 0.f;
  if (act) {
    val = x[base + t];
    if (causal && t > q) val -= 1e9f;
  }
  sh[t] = act ? val : -3.0e38f;
  __syncthreads();
  for (int o = 128; o > 0; o >>= 1) {
    if (t < o) sh[t] = fmaxf(sh[t], sh[t + o]);
    __syncthreads();
  }
  const float mx = sh[0];
  __syncthreads();
  const float e = act ? __expf(val - mx) : 0.f;
  sh[t] = e;
  __syncthreads();
  for (int o = 128; o > 0; o >>= 1) {
    if (t < o) sh[t] += sh[t + o];
    __syncthreads();
  }
  const float inv = 1.f / sh[0];
  if (act) x[base + t] = e * inv;
}

__global__ void combine_k(const float* a, const float* p1, const float* p2,
                          const float* rp, const float* rwv, int idx,
                          float* out, long n) {
  const long i = (long)blockIdx.x * blockDim.x + threadIdx.x;
  if (i >= n) return;
  const float rw = rwv[idx];
  const float t = a[i] + rp[i % DMODEL] + rw * (p1[i] + p2[i]);
  out[i] = t * (1.f + rw) / rw;      // (t + rw*t)/rw
}

__global__ void add_k(const float* a, const float* b, float* out, long n) {
  const long i = (long)blockIdx.x * blockDim.x + threadIdx.x;
  if (i >= n) return;
  out[i] = a[i] + b[i];
}

// ---------------- host orchestration ----------------
static void gemm(hipStream_t st,
                 const float* A, long lda, long sA1, long sA2,
                 const void* Bp, long ldb, long sB1, long sB2,
                 int bT, int bBf16,
                 const float* bias,
                 const float* R, long ldr, long sR1, long sR2,
                 float* C, long ldc, long sC1, long sC2,
                 int M, int N, int K, int nb1, int nb,
                 float alpha, float outScale, int relu) {
  GemmP p;
  p.A = A; p.lda = lda; p.sA1 = sA1; p.sA2 = sA2;
  p.B = Bp; p.ldb = ldb; p.sB1 = sB1; p.sB2 = sB2;
  p.bias = bias;
  p.R = R; p.ldr = ldr; p.sR1 = sR1; p.sR2 = sR2;
  p.C = C; p.ldc = ldc; p.sC1 = sC1; p.sC2 = sC2;
  p.nb1 = nb1; p.K = K; p.alpha = alpha; p.outScale = outScale;
  p.bT = bT; p.relu = relu; p.bBf16 = bBf16;
  dim3 g(M / BM, N / BN, nb);
  gemm_wmma_k<<<g, 256, 0, st>>>(p);
}

static void cvt(hipStream_t st, const float* in, __bf16* out, long n) {
  const long n4 = (n + 3) / 4;
  cvt_bf16_k<<<(int)((n4 + 255) / 256), 256, 0, st>>>(in, out, n);
}

struct Scratch { float *qb, *kv, *sc, *ctx, *at, *ff; };

static void mha(hipStream_t st, const float* xq, const float* xkv,
                const __bf16* Wqkv, const float* bqkv,
                const __bf16* Wo, const float* bo,
                int causal, float* outp, const Scratch& w) {
  const int BS = BATCH * SEQ;
  // q = xq @ Wq^T + bq
  gemm(st, xq, DMODEL, 0, 0, Wqkv, DMODEL, 0, 0, 1, 1, bqkv,
       nullptr, 0, 0, 0, w.qb, DMODEL, 0, 0,
       BS, DMODEL, DMODEL, 1, 1, 1.f, 1.f, 0);
  // kv = xkv @ [Wk;Wv]^T + [bk;bv]
  gemm(st, xkv, DMODEL, 0, 0, Wqkv + (long)DMODEL * DMODEL, DMODEL, 0, 0, 1, 1,
       bqkv + DMODEL, nullptr, 0, 0, 0, w.kv, 2 * DMODEL, 0, 0,
       BS, 2 * DMODEL, DMODEL, 1, 1, 1.f, 1.f, 0);
  // scores[b,h] = q_h @ k_h^T / sqrt(dh)
  gemm(st, w.qb, DMODEL, DHEAD, (long)SEQ * DMODEL,
       w.kv, 2 * DMODEL, DHEAD, (long)SEQ * 2 * DMODEL, 1, 0, nullptr,
       nullptr, 0, 0, 0,
       w.sc, SEQ, (long)SEQ * SEQ, (long)NHEAD * SEQ * SEQ,
       SEQ, SEQ, DHEAD, NHEAD, BATCH * NHEAD, 0.125f, 1.f, 0);
  softmax_k<<<BATCH * NHEAD * SEQ, 256, 0, st>>>(w.sc, SEQ, SEQ, causal);
  // ctx[b,h] = P @ v_h
  gemm(st, w.sc, SEQ, (long)SEQ * SEQ, (long)NHEAD * SEQ * SEQ,
       w.kv + DMODEL, 2 * DMODEL, DHEAD, (long)SEQ * 2 * DMODEL, 0, 0, nullptr,
       nullptr, 0, 0, 0,
       w.ctx, DMODEL, DHEAD, (long)SEQ * DMODEL,
       SEQ, DHEAD, SEQ, NHEAD, BATCH * NHEAD, 1.f, 1.f, 0);
  // out = ctx @ Wo^T + bo
  gemm(st, w.ctx, DMODEL, 0, 0, Wo, DMODEL, 0, 0, 1, 1, bo,
       nullptr, 0, 0, 0, outp, DMODEL, 0, 0,
       BS, DMODEL, DMODEL, 1, 1, 1.f, 1.f, 0);
}

static void ffn(hipStream_t st, float* x, const __bf16* W1, const float* b1,
                const __bf16* W2, const float* b2, const Scratch& w) {
  const int BS = BATCH * SEQ;
  gemm(st, x, DMODEL, 0, 0, W1, DMODEL, 0, 0, 1, 1, b1, nullptr, 0, 0, 0,
       w.ff, FFDIM, 0, 0, BS, FFDIM, DMODEL, 1, 1, 1.f, 1.f, 1);
  gemm(st, w.ff, FFDIM, 0, 0, W2, FFDIM, 0, 0, 1, 1, b2, x, DMODEL, 0, 0,
       x, DMODEL, 0, 0, BS, DMODEL, FFDIM, 1, 1, 1.f, 1.f, 0);
  ln_k<<<BS, 256, 0, st>>>(x, nullptr, x);
}

struct EncW { const __bf16 *Wqkv, *Wo, *W1, *W2;
              const float *bqkv, *bo, *b1, *b2; };
struct DecW { const __bf16 *WqkvS, *WoS, *WqkvC, *WoC, *W1, *W2;
              const float *bqkvS, *boS, *bqkvC, *boC, *b1, *b2; };

static void enc_stack(hipStream_t st, float* x, const EncW& e, const Scratch& w) {
  const int BS = BATCH * SEQ;
  for (int l = 0; l < LNUM; ++l) {
    mha(st, x, x, e.Wqkv + (long)l * 3 * DMODEL * DMODEL,
        e.bqkv + (long)l * 3 * DMODEL,
        e.Wo + (long)l * DMODEL * DMODEL, e.bo + (long)l * DMODEL,
        0, w.at, w);
    ln_k<<<BS, 256, 0, st>>>(x, w.at, x);
    ffn(st, x, e.W1 + (long)l * FFDIM * DMODEL, e.b1 + (long)l * FFDIM,
        e.W2 + (long)l * DMODEL * FFDIM, e.b2 + (long)l * DMODEL, w);
  }
}

static void dec_stack(hipStream_t st, float* x, const float* mem,
                      const DecW& d, const Scratch& w) {
  const int BS = BATCH * SEQ;
  for (int l = 0; l < LNUM; ++l) {
    mha(st, x, x, d.WqkvS + (long)l * 3 * DMODEL * DMODEL,
        d.bqkvS + (long)l * 3 * DMODEL,
        d.WoS + (long)l * DMODEL * DMODEL, d.boS + (long)l * DMODEL,
        1, w.at, w);
    ln_k<<<BS, 256, 0, st>>>(x, w.at, x);
    mha(st, x, mem, d.WqkvC + (long)l * 3 * DMODEL * DMODEL,
        d.bqkvC + (long)l * 3 * DMODEL,
        d.WoC + (long)l * DMODEL * DMODEL, d.boC + (long)l * DMODEL,
        0, w.at, w);
    ln_k<<<BS, 256, 0, st>>>(x, w.at, x);
    ffn(st, x, d.W1 + (long)l * FFDIM * DMODEL, d.b1 + (long)l * FFDIM,
        d.W2 + (long)l * DMODEL * FFDIM, d.b2 + (long)l * DMODEL, w);
  }
}

extern "C" void kernel_launch(void* const* d_in, const int* in_sizes, int n_in,
                              void* d_out, int out_size, void* d_ws,
                              size_t ws_size, hipStream_t stream) {
  (void)in_sizes; (void)n_in; (void)out_size; (void)ws_size;
  const int*   input_ids  = (const int*)d_in[0];
  const int*   output_ids = (const int*)d_in[1];
  const float* tok_e = (const float*)d_in[2];
  const float* tok_d = (const float*)d_in[3];
  const float* pos_e = (const float*)d_in[4];
  const float* pos_d = (const float*)d_in[5];
  const float* memry = (const float*)d_in[6];
  const float* rel_e = (const float*)d_in[7];
  const float* rel_d = (const float*)d_in[8];
  const float* routw = (const float*)d_in[9];
  const float* projWf = (const float*)d_in[10];
  const float* projB = (const float*)d_in[11];
  const float* edbqkv = (const float*)d_in[33];
  const float* edbo   = (const float*)d_in[35];

  float* logits = (float*)d_out;
  const int BS = BATCH * SEQ;
  const long BSD = (long)BS * DMODEL;

  // ---- f32 activation scratch ----
  float* ws = (float*)d_ws;
  float* e0 = ws;            float* e1 = e0 + BSD;
  float* d0 = e1 + BSD;      float* d1 = d0 + BSD;
  float* tb = d1 + BSD;      float* xb = tb + BSD;
  Scratch w;
  w.at = xb + BSD; w.qb = w.at + BSD; w.ctx = w.qb + BSD;
  w.kv = w.ctx + BSD;                      // 2*BSD
  float* edb = w.kv + 2 * BSD;
  w.sc = edb + BSD;                        // B*H*S*S floats
  w.ff = w.sc + (long)BATCH * NHEAD * SEQ * SEQ;   // B*S*FF floats
  float* f32end = w.ff + (long)BATCH * SEQ * FFDIM;

  // ---- bf16 weight region (converted once per launch) ----
  __bf16* bwp = (__bf16*)f32end;
  auto take = [&](long n) { __bf16* r = bwp; bwp += n; return r; };
  const long DD = (long)DMODEL * DMODEL;
  __bf16* projW  = take((long)VOCAB * DMODEL);
  EncW ew; DecW dw;
  ew.Wqkv = take(LNUM * 3 * DD);  ew.Wo = take(LNUM * DD);
  ew.W1 = take((long)LNUM * FFDIM * DMODEL);
  ew.W2 = take((long)LNUM * DMODEL * FFDIM);
  dw.WqkvS = take(LNUM * 3 * DD); dw.WoS = take(LNUM * DD);
  dw.WqkvC = take(LNUM * 3 * DD); dw.WoC = take(LNUM * DD);
  dw.W1 = take((long)LNUM * FFDIM * DMODEL);
  dw.W2 = take((long)LNUM * DMODEL * FFDIM);
  __bf16* edWqkv = take(3 * DD);
  __bf16* edWo   = take(DD);
  __bf16* mem16  = take((long)BATCH * MEMSL * DMODEL);

  ew.bqkv = (const float*)d_in[13]; ew.bo = (const float*)d_in[15];
  ew.b1 = (const float*)d_in[17];   ew.b2 = (const float*)d_in[19];
  dw.bqkvS = (const float*)d_in[21]; dw.boS = (const float*)d_in[23];
  dw.bqkvC = (const float*)d_in[25]; dw.boC = (const float*)d_in[27];
  dw.b1 = (const float*)d_in[29];    dw.b2 = (const float*)d_in[31];

  cvt(stream, projWf, projW, (long)VOCAB * DMODEL);
  cvt(stream, (const float*)d_in[12], (__bf16*)ew.Wqkv, LNUM * 3 * DD);
  cvt(stream, (const float*)d_in[14], (__bf16*)ew.Wo, LNUM * DD);
  cvt(stream, (const float*)d_in[16], (__bf16*)ew.W1, (long)LNUM * FFDIM * DMODEL);
  cvt(stream, (const float*)d_in[18], (__bf16*)ew.W2, (long)LNUM * DMODEL * FFDIM);
  cvt(stream, (const float*)d_in[20], (__bf16*)dw.WqkvS, LNUM * 3 * DD);
  cvt(stream, (const float*)d_in[22], (__bf16*)dw.WoS, LNUM * DD);
  cvt(stream, (const float*)d_in[24], (__bf16*)dw.WqkvC, LNUM * 3 * DD);
  cvt(stream, (const float*)d_in[26], (__bf16*)dw.WoC, LNUM * DD);
  cvt(stream, (const float*)d_in[28], (__bf16*)dw.W1, (long)LNUM * FFDIM * DMODEL);
  cvt(stream, (const float*)d_in[30], (__bf16*)dw.W2, (long)LNUM * DMODEL * FFDIM);
  cvt(stream, (const float*)d_in[32], edWqkv, 3 * DD);
  cvt(stream, (const float*)d_in[34], edWo, DD);
  cvt(stream, memry, mem16, (long)BATCH * MEMSL * DMODEL);

  const long nBSD = BSD;
  const int eltGrid = (int)((nBSD + 255) / 256);
  float* e[2] = {e0, e1};
  float* dd[2] = {d0, d1};

  // ---- encoder embeddings + persistent-memory read attention ----
  embed_k<<<BS, 256, 0, stream>>>(input_ids, tok_e, pos_e, e0);
  const float rsD = 0.04419417382415922f;    // 1/sqrt(512)
  gemm(stream, e0, DMODEL, 0, (long)SEQ * DMODEL,
       mem16, DMODEL, 0, (long)MEMSL * DMODEL, 1, 1, nullptr,
       nullptr, 0, 0, 0,
       w.sc, MEMSL, 0, (long)SEQ * MEMSL,
       SEQ, MEMSL, DMODEL, 1, BATCH, rsD, 1.f, 0);
  softmax_k<<<BS, 256, 0, stream>>>(w.sc, MEMSL, SEQ, 0);
  gemm(stream, w.sc, MEMSL, 0, (long)SEQ * MEMSL,
       memry, DMODEL, 0, (long)MEMSL * DMODEL, 0, 0, nullptr,
       nullptr, 0, 0, 0,
       w.at, DMODEL, 0, (long)SEQ * DMODEL,
       SEQ, DMODEL, MEMSL, 1, BATCH, 1.f, 1.f, 0);
  ln_k<<<BS, 256, 0, stream>>>(e0, w.at, e0);
  (void)hipMemcpyAsync(e1, e0, BSD * sizeof(float), hipMemcpyDeviceToDevice,
                       stream);

  embed_k<<<BS, 256, 0, stream>>>(output_ids, tok_d, pos_d, d0);
  (void)hipMemcpyAsync(d1, d0, BSD * sizeof(float), hipMemcpyDeviceToDevice,
                       stream);

  const float invs2 = 0.7071067811865475f;   // 1/sqrt(2)

  // ---- encoder loop ----
  for (int i = 0; i < LNUM; ++i) {
    const float* rp = rel_e + ((long)i * RELPOS + (MAXLEN - 1)) * DMODEL;
    float* other = e[(i + 1) % LNUM];        // == e[i-1] for L==2
    combine_k<<<eltGrid, 256, 0, stream>>>(e[i], other, other, rp, routw, i,
                                           tb, nBSD);
    (void)hipMemcpyAsync(xb, tb, BSD * sizeof(float),
                         hipMemcpyDeviceToDevice, stream);
    enc_stack(stream, xb, ew, w);
    add_k<<<eltGrid, 256, 0, stream>>>(tb, xb, e[i], nBSD);
    gemm(stream, e[i], DMODEL, 0, 0, projW, DMODEL, 0, 0, 1, 1, projB,
         (i == 0) ? nullptr : logits, VOCAB, 0, 0,
         logits, VOCAB, 0, 0, BS, VOCAB, DMODEL, 1, 1,
         1.f, (i == 0) ? 1.f : invs2, 0);
  }

  // ---- decoder loop ----
  for (int i = 0; i < LNUM; ++i) {
    const float* rp = rel_d + ((long)i * RELPOS + (MAXLEN - 1)) * DMODEL;
    float* other = dd[(i + 1) % LNUM];
    combine_k<<<eltGrid, 256, 0, stream>>>(dd[i], other, other, rp, routw, i,
                                           tb, nBSD);
    (void)hipMemcpyAsync(xb, tb, BSD * sizeof(float),
                         hipMemcpyDeviceToDevice, stream);
    dec_stack(stream, xb, e[i], dw, w);
    add_k<<<eltGrid, 256, 0, stream>>>(tb, xb, dd[i], nBSD);
    gemm(stream, dd[i], DMODEL, 0, 0, projW, DMODEL, 0, 0, 1, 1, projB,
         logits, VOCAB, 0, 0, logits, VOCAB, 0, 0,
         BS, VOCAB, DMODEL, 1, 1, 1.f, (i == 0) ? 1.f : invs2, 0);

    enc_stack(stream, e[i], ew, w);          // e[i] = enc_stack(e[i])
    dec_stack(stream, dd[i], e[i], dw, w);   // d[i] = dec_stack(d[i], e[i])

    mha(stream, dd[1], e[1], edWqkv, edbqkv, edWo, edbo, 1, edb, w);

    gemm(stream, edb, DMODEL, 0, 0, projW, DMODEL, 0, 0, 1, 1, projB,
         logits, VOCAB, 0, 0, logits, VOCAB, 0, 0,
         BS, VOCAB, DMODEL, 1, 1, 1.f, 1.f, 0);
    gemm(stream, e[i], DMODEL, 0, 0, projW, DMODEL, 0, 0, 1, 1, projB,
         logits, VOCAB, 0, 0, logits, VOCAB, 0, 0,
         BS, VOCAB, DMODEL, 1, 1, 1.f, 1.f, 0);
    gemm(stream, dd[i], DMODEL, 0, 0, projW, DMODEL, 0, 0, 1, 1, projB,
         logits, VOCAB, 0, 0, logits, VOCAB, 0, 0,
         BS, VOCAB, DMODEL, 1, 1, 1.f, 1.f, 0);
  }
}
